// CodeBook_19705309954686
// MI455X (gfx1250) — compile-verified, matrix operand order checked by reading, not run on hardware
//
#include <hip/hip_runtime.h>

// ---------------------------------------------------------------------------
// VQ codebook lookup for MI455X (gfx1250, wave32).
//   distances[N,K] = |e|^2 - 2 x.e   (|x|^2 row-constant, dropped for argmin)
//   x.e computed with bf16 hi/lo split WMMAs (3x v_wmma_f32_16x16x32_bf16 per
//   32-wide K chunk) for ~fp32 accuracy at matrix-core rate.
// Outputs (flat, in reference return order):
//   out[0]                      loss = 1.25 * mean((E[idx]-x)^2)
//   out[1 .. N*D]               quantized_st (== E[idx] numerically)
//   out[1+N*D .. 1+N*D+N*K-1]   one-hot encodings
// ---------------------------------------------------------------------------

typedef __attribute__((ext_vector_type(16))) __bf16 v16bf;
typedef __attribute__((ext_vector_type(8)))  float  v8f;

#define N_ROWS  131072
#define K_CODES 1024
#define D_DIM   128

// ---------------- stage 1: codebook bf16 hi/lo split + row norms -----------
__global__ void vq_prep(const float* __restrict__ E,
                        __bf16* __restrict__ ehi, __bf16* __restrict__ elo,
                        float* __restrict__ enorm)
{
    __shared__ float s[128];
    const int k = blockIdx.x, d = threadIdx.x;
    float e = E[(size_t)k * D_DIM + d];
    __bf16 hi = (__bf16)e;
    ehi[(size_t)k * D_DIM + d] = hi;
    elo[(size_t)k * D_DIM + d] = (__bf16)(e - (float)hi);
    s[d] = e * e;
    __syncthreads();
    for (int st = 64; st > 0; st >>= 1) {
        if (d < st) s[d] += s[d + st];
        __syncthreads();
    }
    if (d == 0) enorm[k] = s[0];
}

// ---------------- stage 2: distances + argmin + outputs --------------------
// 256 threads = 8 wave32; each wave owns a 16-row strip; block owns 128 rows.
__global__ void __launch_bounds__(256)
vq_main(const float* __restrict__ X, const float* __restrict__ E,
        const __bf16* __restrict__ ehi, const __bf16* __restrict__ elo,
        const float* __restrict__ enorm,
        float* __restrict__ out, float* __restrict__ partial)
{
    __shared__ float s_enorm[K_CODES];
    __shared__ int   s_idx[128];
    __shared__ float s_red[256];

    const int tid = threadIdx.x;
    for (int i = tid; i < K_CODES; i += 256) s_enorm[i] = enorm[i];

    const int wave = tid >> 5;
    const int lane = tid & 31;
    const int m    = lane & 15;   // row-in-tile (A) / col-in-tile (B, C/D)
    const int h    = lane >> 4;   // half-select per ISA 16-bit fragment layout
    const size_t g = (size_t)blockIdx.x * 128 + wave * 16 + m;

    // A-fragments: 16x32 bf16 per chunk; lane holds K runs [kc+8h, +8) and
    // [kc+16+8h, +8) per the ISA A-matrix table. Split fp32 -> bf16 hi/lo.
    v16bf ahi[4], alo[4];
    const float4* xr = (const float4*)(X + g * D_DIM);
    #pragma unroll
    for (int c = 0; c < 4; ++c) {
        float4 f0 = xr[8*c + 2*h],     f1 = xr[8*c + 2*h + 1];
        float4 f2 = xr[8*c + 4 + 2*h], f3 = xr[8*c + 4 + 2*h + 1];
        float xv[16] = { f0.x,f0.y,f0.z,f0.w, f1.x,f1.y,f1.z,f1.w,
                         f2.x,f2.y,f2.z,f2.w, f3.x,f3.y,f3.z,f3.w };
        #pragma unroll
        for (int i = 0; i < 16; ++i) {
            __bf16 hi = (__bf16)xv[i];
            ahi[c][i] = hi;
            alo[c][i] = (__bf16)(xv[i] - (float)hi);
        }
    }
    __syncthreads();

    float best[8];
    int   bidx[8];
    #pragma unroll
    for (int r = 0; r < 8; ++r) { best[r] = 3.0e38f; bidx[r] = 0; }

    union BFrag { uint4 u[2]; v16bf v; };

    #pragma unroll 1
    for (int t = 0; t < 64; ++t) {
        const int code = t * 16 + m;     // this lane's codebook column
        const uint4* bh = (const uint4*)(ehi + (size_t)code * D_DIM);
        const uint4* bl = (const uint4*)(elo + (size_t)code * D_DIM);
        if (t < 63) {   // pull next tile toward the WGP (global_prefetch_b8)
            __builtin_prefetch(ehi + (size_t)(code + 16) * D_DIM, 0, 3);
            __builtin_prefetch(elo + (size_t)(code + 16) * D_DIM, 0, 3);
        }
        v8f acc = {};
        #pragma unroll
        for (int c = 0; c < 4; ++c) {
            BFrag fh, fl;                       // B mirrors A's lane packing
            fh.u[0] = bh[4*c + h];  fh.u[1] = bh[4*c + 2 + h];
            fl.u[0] = bl[4*c + h];  fl.u[1] = bl[4*c + 2 + h];
            acc = __builtin_amdgcn_wmma_f32_16x16x32_bf16(
                      false, ahi[c], false, fh.v, (short)0, acc, false, false);
            acc = __builtin_amdgcn_wmma_f32_16x16x32_bf16(
                      false, ahi[c], false, fl.v, (short)0, acc, false, false);
            acc = __builtin_amdgcn_wmma_f32_16x16x32_bf16(
                      false, alo[c], false, fh.v, (short)0, acc, false, false);
        }
        const float en = s_enorm[code];
        #pragma unroll
        for (int r = 0; r < 8; ++r) {           // C/D layout: VGPR r = row r+8h
            float dist = en - 2.0f * acc[r];
            if (dist < best[r]) { best[r] = dist; bidx[r] = code; }
        }
    }

    // cross-lane argmin within each 16-lane half (rows M = r + 8h)
    #pragma unroll
    for (int r = 0; r < 8; ++r) {
        float d = best[r]; int i = bidx[r];
        #pragma unroll
        for (int msk = 1; msk < 16; msk <<= 1) {
            float od = __shfl_xor(d, msk, 32);
            int   oi = __shfl_xor(i, msk, 32);
            if (od < d || (od == d && oi < i)) { d = od; i = oi; }
        }
        if (m == 0) s_idx[wave * 16 + r + 8 * h] = i;
    }
    __syncthreads();

    const size_t QBASE = 1;
    const size_t EBASE = 1 + (size_t)N_ROWS * D_DIM;

    // quantized rows (== straight-through value) + loss partial
    {
        int row = tid >> 1, half = tid & 1;
        size_t grow = (size_t)blockIdx.x * 128 + row;
        int idx = s_idx[row];
        const float* qsrc = E + (size_t)idx * D_DIM + half * 64;
        const float* xsrc = X + grow * D_DIM + half * 64;
        float* qdst = out + QBASE + grow * D_DIM + half * 64;
        float lsum = 0.f;
        #pragma unroll 4
        for (int d2 = 0; d2 < 64; ++d2) {
            float q = qsrc[d2], x = xsrc[d2];
            qdst[d2] = q;
            float df = q - x;
            lsum += df * df;
        }
        s_red[tid] = lsum;
    }

    // one-hot encodings: each thread zeroes half a row then sets its 1.0
    {
        int row = tid >> 1, half = tid & 1;
        size_t grow = (size_t)blockIdx.x * 128 + row;
        int idx = s_idx[row];
        float* enc = out + EBASE + grow * (size_t)K_CODES + half * 512;
        // element offset == 1 (mod 4): 3 scalars + 127 aligned float4 + 1 scalar
        enc[0] = 0.f; enc[1] = 0.f; enc[2] = 0.f;
        float4* v4 = (float4*)(enc + 3);
        float4 z = {0.f, 0.f, 0.f, 0.f};
        #pragma unroll 4
        for (int i = 0; i < 127; ++i) v4[i] = z;
        enc[511] = 0.f;
        int cbase = half * 512;
        if (idx >= cbase && idx < cbase + 512) enc[idx - cbase] = 1.0f;
    }

    __syncthreads();
    for (int st = 128; st > 0; st >>= 1) {
        if (tid < st) s_red[tid] += s_red[tid + st];
        __syncthreads();
    }
    if (tid == 0) partial[blockIdx.x] = s_red[0];
}

// ---------------- stage 3: deterministic loss reduction --------------------
__global__ void vq_reduce(const float* __restrict__ partial, float* __restrict__ out)
{
    __shared__ float s[256];
    int t = threadIdx.x;
    float v = partial[t] + partial[t + 256] + partial[t + 512] + partial[t + 768];
    s[t] = v;
    __syncthreads();
    for (int st = 128; st > 0; st >>= 1) {
        if (t < st) s[t] += s[t + st];
        __syncthreads();
    }
    // loss = q_latent + 0.25*e_latent = 1.25 * mean((E[idx]-x)^2)
    if (t == 0) out[0] = s[0] * (1.25f / 16777216.0f);
}

// ---------------------------------------------------------------------------
extern "C" void kernel_launch(void* const* d_in, const int* in_sizes, int n_in,
                              void* d_out, int out_size, void* d_ws, size_t ws_size,
                              hipStream_t stream)
{
    const float* X = (const float*)d_in[0];   // c_input [N, D]
    const float* E = (const float*)d_in[1];   // embedding_weight [K, D]
    float* out = (float*)d_out;

    char* ws = (char*)d_ws;
    __bf16* ehi   = (__bf16*)(ws);                       // K*D*2 = 256 KB
    __bf16* elo   = (__bf16*)(ws + 262144);              // 256 KB
    float*  enorm = (float*) (ws + 524288);              // 4 KB
    float*  part  = (float*) (ws + 528384);              // 4 KB

    vq_prep  <<<K_CODES,      128, 0, stream>>>(E, ehi, elo, enorm);
    vq_main  <<<N_ROWS / 128, 256, 0, stream>>>(X, E, ehi, elo, enorm, out, part);
    vq_reduce<<<1,            256, 0, stream>>>(part, out);
}
